// GNNEncoder_90151363543321
// MI455X (gfx1250) — compile-verified, matrix operand order checked by reading, not run on hardware
//
#include <hip/hip_runtime.h>

typedef __attribute__((ext_vector_type(16))) __bf16 v16bf;
typedef __attribute__((ext_vector_type(8)))  float  v8f;

union BF16Frag { v16bf v; __bf16 h[16]; unsigned short u[16]; unsigned int w[8]; };

__device__ __forceinline__ unsigned short f2bf(float f) {
    union { __bf16 b; unsigned short u; } cv;
    cv.b = (__bf16)f;   // hardware f32->bf16 convert (RNE)
    return cv.u;
}

// ---------------------------------------------------------------------------
// elementwise / scatter kernels
// ---------------------------------------------------------------------------
__global__ __launch_bounds__(256)
void zero_kernel(float* __restrict__ p, int n) {
    int i = blockIdx.x * 256 + threadIdx.x;
    if (i < n) p[i] = 0.0f;
}

__global__ __launch_bounds__(256)
void copy_kernel(float* __restrict__ dst, const float* __restrict__ src, int n) {
    int i = blockIdx.x * 256 + threadIdx.x;
    if (i < n) dst[i] = src[i];
}

// one wave per edge: 32 lanes x float4 = 128 features
__global__ __launch_bounds__(256)
void scatter_kernel(const float* __restrict__ x, const int* __restrict__ ei,
                    float* __restrict__ agg, int E) {
    const int wid  = (blockIdx.x * 256 + threadIdx.x) >> 5;
    const int lane = threadIdx.x & 31;
    if (wid >= E) return;
    const int src = ei[wid];
    const int dst = ei[E + wid];
    const float4 v = ((const float4*)(x + (size_t)src * 128))[lane];
    float* a = agg + (size_t)dst * 128 + lane * 4;
    atomicAdd(a + 0, v.x);
    atomicAdd(a + 1, v.y);
    atomicAdd(a + 2, v.z);
    atomicAdd(a + 3, v.w);
}

__device__ __forceinline__ void pack4(BF16Frag& f, int o, float4 p) {
    f.u[o + 0] = f2bf(p.x);
    f.u[o + 1] = f2bf(p.y);
    f.u[o + 2] = f2bf(p.z);
    f.u[o + 3] = f2bf(p.w);
}

// ---------------------------------------------------------------------------
// Fused GIN MLP: H = relu(X@WA + BA) @ WB + BB   (bf16 WMMA, f32 accumulate)
// X and H are padded to a multiple of 128 rows -> all tile loads/stores are
// unconditional (branch-free); BN stats mask out padded rows via select.
// 256 threads = 8 waves; each wave owns a 16-row tile; block = 128 rows.
// ---------------------------------------------------------------------------
template<int NT2>   // number of 16-wide output column tiles (8 -> 128, 4 -> 64)
__global__ __launch_bounds__(256)
void gin_mlp_kernel(const float* __restrict__ X,
                    const float* __restrict__ WA, const float* __restrict__ BA,
                    const float* __restrict__ WB, const float* __restrict__ BB,
                    float* __restrict__ H,
                    float* __restrict__ colsum, float* __restrict__ colsq,
                    int n) {
    constexpr int DOUT = NT2 * 16;
    __shared__ unsigned short sW[128 * 128];      // weights, transposed [n][k], bf16
    __shared__ unsigned short sInt[8 * 16 * 128]; // per-wave intermediate, bf16

    const int tid  = threadIdx.x;
    const int wave = tid >> 5;
    const int lane = tid & 31;
    const int half = lane >> 4;
    const int m    = lane & 15;

    // stage WA^T into LDS (k contiguous per output column -> ds_load_b128 B-frags)
    for (int idx = tid; idx < 128 * 128; idx += 256) {
        const int nn = idx >> 7, kk = idx & 127;
        sW[idx] = f2bf(WA[kk * 128 + nn]);
    }
    __syncthreads();

    const int rowBase = blockIdx.x * 128 + wave * 16;
    const int arow    = rowBase + m;                 // A-fragment row (padded, in-bounds)
    const float4* xr4 = (const float4*)(X + (size_t)arow * 128);

    const v8f vzero = {};
    v8f acc1[8];
#pragma unroll
    for (int t = 0; t < 8; ++t) acc1[t] = vzero;

#pragma unroll
    for (int k0 = 0; k0 < 128; k0 += 32) {
        BF16Frag a;
        const int q0 = (k0 + half * 8) >> 2;         // float4 index of k-group 0
        const int q1 = (k0 + 16 + half * 8) >> 2;    // float4 index of k-group 1
        pack4(a, 0,  xr4[q0]);
        pack4(a, 4,  xr4[q0 + 1]);
        pack4(a, 8,  xr4[q1]);
        pack4(a, 12, xr4[q1 + 1]);
#pragma unroll
        for (int t = 0; t < 8; ++t) {
            BF16Frag b;
            const unsigned short* wp = &sW[(t * 16 + m) * 128];
#pragma unroll
            for (int j = 0; j < 8; ++j) {
                const int k = k0 + 2 * j + ((j < 4) ? 0 : 8) + half * 8;
                b.w[j] = *(const unsigned int*)(wp + k);
            }
            acc1[t] = __builtin_amdgcn_wmma_f32_16x16x32_bf16(
                false, a.v, false, b.v, (short)0, acc1[t], false, false);
        }
    }

    // bias + relu, stash intermediate (row-major 16x128 bf16) in wave-private LDS
    unsigned short* ip = &sInt[wave * 16 * 128];
#pragma unroll
    for (int t = 0; t < 8; ++t) {
        const int c = t * 16 + m;
        const float bav = BA[c];
#pragma unroll
        for (int r = 0; r < 8; ++r) {
            const int rl = r + half * 8;
            float v = acc1[t][r] + bav;
            v = v > 0.0f ? v : 0.0f;
            ip[rl * 128 + c] = f2bf(v);
        }
    }

    __syncthreads();   // all waves done reading WA from sW
    for (int idx = tid; idx < DOUT * 128; idx += 256) {
        const int nn = idx >> 7, kk = idx & 127;
        sW[idx] = f2bf(WB[kk * DOUT + nn]);
    }
    __syncthreads();

    v8f acc2[NT2];
#pragma unroll
    for (int t = 0; t < NT2; ++t) acc2[t] = vzero;

#pragma unroll
    for (int k0 = 0; k0 < 128; k0 += 32) {
        BF16Frag a;
#pragma unroll
        for (int j = 0; j < 8; ++j) {
            const int k = k0 + 2 * j + ((j < 4) ? 0 : 8) + half * 8;
            a.w[j] = *(const unsigned int*)(ip + m * 128 + k);
        }
#pragma unroll
        for (int t = 0; t < NT2; ++t) {
            BF16Frag b;
            const unsigned short* wp = &sW[(t * 16 + m) * 128];
#pragma unroll
            for (int j = 0; j < 8; ++j) {
                const int k = k0 + 2 * j + ((j < 4) ? 0 : 8) + half * 8;
                b.w[j] = *(const unsigned int*)(wp + k);
            }
            acc2[t] = __builtin_amdgcn_wmma_f32_16x16x32_bf16(
                false, a.v, false, b.v, (short)0, acc2[t], false, false);
        }
    }

    // epilogue: bias, unconditional store into padded H, masked BN stats
#pragma unroll
    for (int t = 0; t < NT2; ++t) {
        const int c = t * 16 + m;
        const float bbv = BB[c];
        float s = 0.0f, sq = 0.0f;
#pragma unroll
        for (int r = 0; r < 8; ++r) {
            const int grow = rowBase + r + half * 8;
            const float v = acc2[t][r] + bbv;
            H[(size_t)grow * DOUT + c] = v;
            const float vm = (grow < n) ? v : 0.0f;   // branch-free select
            s  += vm;
            sq += vm * vm;
        }
        atomicAdd(&colsum[c], s);
        atomicAdd(&colsq[c],  sq);
    }
}

// ---------------------------------------------------------------------------
// BatchNorm finalize + apply, pooling
// ---------------------------------------------------------------------------
__global__ void bn_finalize_kernel(const float* __restrict__ colsum,
                                   const float* __restrict__ colsq,
                                   const float* __restrict__ g,
                                   const float* __restrict__ be,
                                   float* __restrict__ cA, float* __restrict__ cB,
                                   int dout, float invN) {
    const int c = threadIdx.x;
    if (c < dout) {
        const float mu  = colsum[c] * invN;
        const float var = colsq[c] * invN - mu * mu;
        const float a   = g[c] * rsqrtf(var + 1e-5f);
        cA[c] = a;
        cB[c] = be[c] - mu * a;
    }
}

// y = relu(h*a + b); writes next-layer features AND pre-initializes next AGG
__global__ __launch_bounds__(256)
void bn_apply_kernel(const float* __restrict__ H,
                     const float* __restrict__ cA, const float* __restrict__ cB,
                     float* __restrict__ xo, float* __restrict__ aggo, int total) {
    const int i = blockIdx.x * 256 + threadIdx.x;
    if (i >= total) return;
    const int c = i & 127;
    float v = H[i] * cA[c] + cB[c];
    v = v > 0.0f ? v : 0.0f;
    xo[i]   = v;
    aggo[i] = v;
}

__global__ __launch_bounds__(256)
void bn_apply_pool_kernel(const float* __restrict__ H,
                          const float* __restrict__ cA, const float* __restrict__ cB,
                          const int* __restrict__ batch,
                          float* __restrict__ sums, int n) {
    const int i = blockIdx.x * 256 + threadIdx.x;
    if (i >= n * 64) return;
    const int c    = i & 63;
    const int node = i >> 6;
    float v = H[i] * cA[c] + cB[c];
    v = v > 0.0f ? v : 0.0f;
    atomicAdd(&sums[(size_t)batch[node] * 64 + c], v);
}

__global__ __launch_bounds__(256)
void count_kernel(const int* __restrict__ batch, float* __restrict__ cnt, int n) {
    const int i = blockIdx.x * 256 + threadIdx.x;
    if (i < n) atomicAdd(&cnt[batch[i]], 1.0f);
}

__global__ __launch_bounds__(256)
void pool_out_kernel(const float* __restrict__ sums, const float* __restrict__ cnt,
                     float* __restrict__ out, int total) {
    const int i = blockIdx.x * 256 + threadIdx.x;
    if (i >= total) return;
    const int g = i >> 6;
    out[i] = sums[i] / fmaxf(cnt[g], 1.0f);
}

// ---------------------------------------------------------------------------
extern "C" void kernel_launch(void* const* d_in, const int* in_sizes, int n_in,
                              void* d_out, int out_size, void* d_ws, size_t ws_size,
                              hipStream_t stream) {
    (void)n_in; (void)out_size; (void)ws_size;
    const float* x     = (const float*)d_in[0];
    const int*   ei    = (const int*)d_in[1];
    const int*   batch = (const int*)d_in[2];
    const int N = in_sizes[0] / 128;
    const int E = in_sizes[1] / 2;

    const float *WA_[3], *BA_[3], *WB_[3], *BB_[3], *G_[3], *BE_[3];
    for (int l = 0; l < 3; ++l) {
        const int b = 3 + l * 6;
        WA_[l] = (const float*)d_in[b + 0];
        BA_[l] = (const float*)d_in[b + 1];
        WB_[l] = (const float*)d_in[b + 2];
        BB_[l] = (const float*)d_in[b + 3];
        G_[l]  = (const float*)d_in[b + 4];
        BE_[l] = (const float*)d_in[b + 5];
    }

    const int gemmBlocks = (N + 127) / 128;
    const size_t NPAD = (size_t)gemmBlocks * 128;    // rows padded to tile grid

    char* ws = (char*)d_ws;
    size_t off = 0;
    float* XBUF = (float*)(ws + off); off += NPAD * 128 * sizeof(float);
    float* AGG  = (float*)(ws + off); off += NPAD * 128 * sizeof(float);
    float* HPRE = (float*)(ws + off); off += NPAD * 128 * sizeof(float);
    float* colsum = (float*)(ws + off); off += 128 * sizeof(float);
    float* colsq  = (float*)(ws + off); off += 128 * sizeof(float);
    float* cA = (float*)(ws + off); off += 128 * sizeof(float);
    float* cB = (float*)(ws + off); off += 128 * sizeof(float);
    float* psum = (float*)(ws + off); off += 128 * 64 * sizeof(float);
    float* pcnt = (float*)(ws + off); off += 128 * sizeof(float);

    const int elemBlocks = (N * 128 + 255) / 256;
    const int edgeBlocks = (E + 7) / 8;
    const float invN = 1.0f / (float)N;

    // ---------------- layer 0 ----------------
    zero_kernel<<<1, 256, 0, stream>>>(colsum, 256);              // colsum + colsq
    copy_kernel<<<elemBlocks, 256, 0, stream>>>(AGG, x, N * 128);
    scatter_kernel<<<edgeBlocks, 256, 0, stream>>>(x, ei, AGG, E);
    gin_mlp_kernel<8><<<gemmBlocks, 256, 0, stream>>>(AGG, WA_[0], BA_[0], WB_[0], BB_[0],
                                                      HPRE, colsum, colsq, N);
    bn_finalize_kernel<<<1, 128, 0, stream>>>(colsum, colsq, G_[0], BE_[0], cA, cB, 128, invN);
    bn_apply_kernel<<<elemBlocks, 256, 0, stream>>>(HPRE, cA, cB, XBUF, AGG, N * 128);

    // ---------------- layer 1 ----------------
    zero_kernel<<<1, 256, 0, stream>>>(colsum, 256);
    scatter_kernel<<<edgeBlocks, 256, 0, stream>>>(XBUF, ei, AGG, E);
    gin_mlp_kernel<8><<<gemmBlocks, 256, 0, stream>>>(AGG, WA_[1], BA_[1], WB_[1], BB_[1],
                                                      HPRE, colsum, colsq, N);
    bn_finalize_kernel<<<1, 128, 0, stream>>>(colsum, colsq, G_[1], BE_[1], cA, cB, 128, invN);
    bn_apply_kernel<<<elemBlocks, 256, 0, stream>>>(HPRE, cA, cB, XBUF, AGG, N * 128);

    // ---------------- layer 2 ----------------
    zero_kernel<<<1, 256, 0, stream>>>(colsum, 256);
    scatter_kernel<<<edgeBlocks, 256, 0, stream>>>(XBUF, ei, AGG, E);
    gin_mlp_kernel<4><<<gemmBlocks, 256, 0, stream>>>(AGG, WA_[2], BA_[2], WB_[2], BB_[2],
                                                      HPRE, colsum, colsq, N);
    bn_finalize_kernel<<<1, 64, 0, stream>>>(colsum, colsq, G_[2], BE_[2], cA, cB, 64, invN);

    // pooled mean per graph
    zero_kernel<<<(128 * 64 + 128 + 255) / 256, 256, 0, stream>>>(psum, 128 * 64 + 128); // psum+pcnt
    bn_apply_pool_kernel<<<(N * 64 + 255) / 256, 256, 0, stream>>>(HPRE, cA, cB, batch, psum, N);
    count_kernel<<<(N + 255) / 256, 256, 0, stream>>>(batch, pcnt, N);
    pool_out_kernel<<<(128 * 64 + 255) / 256, 256, 0, stream>>>(psum, pcnt, (float*)d_out, 128 * 64);
}
